// PointnetLFPModuleMSG_3169685865256
// MI455X (gfx1250) — matrix operand order repaired and checked by reading, not compiled, not run on hardware
//
#include <hip/hip_runtime.h>

// ---------------------------------------------------------------------------
// PointNet++ LFP MSG module for MI455X (gfx1250, wave32, WMMA + async LDS)
// B=4, N1=8192, N2=2048, C1=C2=64, radii (0.2,0.4), nsamples (16,32)
// ---------------------------------------------------------------------------

static constexpr int B_  = 4;
static constexpr int N1_ = 8192;
static constexpr int N2_ = 2048;

typedef __attribute__((ext_vector_type(16))) _Float16 v16h;
typedef __attribute__((ext_vector_type(8)))  float    v8f;

union V16 { v16h v; uint4 u[2]; };   // 32-byte operand view

static __device__ __forceinline__ void lds_fence() {
    asm volatile("s_wait_dscnt 0x0" ::: "memory");
}

// Async global->LDS 16B copy (CDNA5, tracked by ASYNCcnt).
// VDST = VGPR holding LDS byte address (low 32 bits of the flat address,
// per aperture mapping ISA 10.2), VADDR = 64-bit global address.
static __device__ __forceinline__ void async_copy_b128(const _Float16* lds_dst,
                                                       const _Float16* gsrc) {
    unsigned l = (unsigned)(size_t)lds_dst;
    asm volatile("global_load_async_to_lds_b128 %0, %1, off"
                 :: "v"(l), "v"(gsrc) : "memory");
}
static __device__ __forceinline__ void async_wait() {
    asm volatile("s_wait_asynccnt 0x0" ::: "memory");
}

// combine lane l with lane l^16 via LDS swizzle (SWAPX16: xor=0x10,and=0x1f)
static __device__ __forceinline__ float swap16_max(float v) {
    int s = __builtin_amdgcn_ds_swizzle(__float_as_int(v), 0x401F);
    return fmaxf(v, __int_as_float(s));
}

static __device__ __forceinline__ v8f wmma_f16(v16h a, v16h b, v8f c) {
    return __builtin_amdgcn_wmma_f32_16x16x32_f16(
        /*neg_a=*/false, a, /*neg_b=*/false, b,
        /*c_mod=*/(short)0, c, /*reuse_a=*/false, /*reuse_b=*/false);
}

// contiguous per-lane operand load: lane's 16 halves at p + lane*16
static __device__ __forceinline__ v16h ld_op(const _Float16* p, int lane) {
    return *(const v16h*)(p + lane * 16);
}

// Operand-major ("swizzled") layouts:
// A operand (16x32 f16): m = lane%16, k(h) = (h/8)*16 + (lane/16)*8 + (h%8)
// B operand (32x16 f16): n = lane%16, k(h) = (h/8)*16 + (lane/16)*8 + (h%8)
// Storage: [tile][lane][h] contiguous, 32B per lane per operand.

static constexpr int W0SZ = 3 * 4 * 32 * 16;   // 6144 halves per scale
static constexpr int W1SZ = 2 * 8 * 32 * 16;   // 8192 halves per scale
static constexpr int WPSZ = 6 * 8 * 32 * 16;   // 24576 halves

// ---------------------------------------------------------------------------
// Kernel 1: ball query, both radii in one ordered wave32 scan per query.
// ---------------------------------------------------------------------------
__global__ __launch_bounds__(128) void bq_kernel(
    const float* __restrict__ xyz2, const float* __restrict__ xyz1,
    int* __restrict__ idx16, int* __restrict__ idx32)
{
    const int lane = threadIdx.x & 31;
    const int qi = blockIdx.x * 4 + (threadIdx.x >> 5);   // flat b*N2+n2
    const int b = qi >> 11;

    const float qx = xyz2[qi * 3 + 0];
    const float qy = xyz2[qi * 3 + 1];
    const float qz = xyz2[qi * 3 + 2];
    const float* p1 = xyz1 + (size_t)b * N1_ * 3;

    const float r0 = 0.2f, r1 = 0.4f;
    const float r0sq = r0 * r0, r1sq = r1 * r1;

    int cnt16 = 0, cnt32 = 0, first16 = 0, first32 = 0;
    const unsigned lanemask = (1u << lane) - 1u;

    for (int j0 = 0; j0 < N1_; j0 += 32) {
        __builtin_prefetch(p1 + 3 * (j0 + 128));
        const int j = j0 + lane;
        const float dx = p1[j * 3 + 0] - qx;
        const float dy = p1[j * 3 + 1] - qy;
        const float dz = p1[j * 3 + 2] - qz;
        const float d2 = dx * dx + dy * dy + dz * dz;
        const unsigned m1 = __builtin_amdgcn_ballot_w32(d2 <= r1sq);
        const unsigned m0 = __builtin_amdgcn_ballot_w32(d2 <= r0sq);
        if (m1) {
            if (cnt32 == 0) first32 = j0 + __ffs(m1) - 1;
            if ((m1 >> lane) & 1u) {
                int pos = cnt32 + __popc(m1 & lanemask);
                if (pos < 32) idx32[(size_t)qi * 32 + pos] = j;
            }
            cnt32 += __popc(m1);
        }
        if (m0) {
            if (cnt16 == 0) first16 = j0 + __ffs(m0) - 1;
            if ((m0 >> lane) & 1u) {
                int pos = cnt16 + __popc(m0 & lanemask);
                if (pos < 16) idx16[(size_t)qi * 16 + pos] = j;
            }
            cnt16 += __popc(m0);
        }
        if (cnt16 >= 16 && cnt32 >= 32) break;
    }
    const int pad16 = cnt16 ? first16 : 0;
    const int pad32 = cnt32 ? first32 : 0;
    if (lane < 16) { if (lane >= cnt16) idx16[(size_t)qi * 16 + lane] = pad16; }
    if (lane >= cnt32) idx32[(size_t)qi * 32 + lane] = pad32;
}

// ---------------------------------------------------------------------------
// Transpose (B,C,N) f32 -> (B,N,C) f16 so neighbor gathers are contiguous.
// ---------------------------------------------------------------------------
__global__ void prep_transpose_kernel(const float* __restrict__ in,
                                      _Float16* __restrict__ out,
                                      int C, int N, int total)
{
    int i = blockIdx.x * 256 + threadIdx.x;
    if (i >= total) return;
    int c = i % C;
    int t = i / C;
    int n = t % N;
    int b = t / N;
    out[i] = (_Float16)in[((size_t)b * C + c) * N + n];
}

// ---------------------------------------------------------------------------
// Pre-swizzle ALL weights into operand-major f16 global buffers (one-time).
// W0 logical rows: [0,64) feature w, [64,67) coord w, [67,96) zero.
// ---------------------------------------------------------------------------
__global__ void prep_weights_kernel(
    const float* __restrict__ w00, const float* __restrict__ w01,
    const float* __restrict__ w10, const float* __restrict__ w11,
    const float* __restrict__ wp,
    _Float16* __restrict__ wsw0, _Float16* __restrict__ wsw1,
    _Float16* __restrict__ wswp)
{
    const int tid = blockIdx.x * 256 + threadIdx.x;
    const int nthr = gridDim.x * 256;
    for (int e = tid; e < 2 * 96 * 64; e += nthr) {
        int s = e / (96 * 64), r = e % (96 * 64);
        int k = r >> 6, nn = r & 63;
        const float* w0 = s ? w10 : w00;
        float v = 0.f;
        if (k < 64)      v = w0[nn * 67 + 3 + k];
        else if (k < 67) v = w0[nn * 67 + (k - 64)];
        int kt = k >> 5, kk = k & 31;
        int lan = (nn & 15) + (((kk >> 3) & 1) << 4);
        wsw0[s * W0SZ + ((kt * 4 + (nn >> 4)) * 32 + lan) * 16 + ((kk & 7) + ((kk >> 4) << 3))] =
            (_Float16)v;
    }
    for (int e = tid; e < 2 * 64 * 128; e += nthr) {
        int s = e / (64 * 128), r = e % (64 * 128);
        int k = r >> 7, nn = r & 127;
        const float* w1 = s ? w11 : w01;
        int kt = k >> 5, kk = k & 31;
        int lan = (nn & 15) + (((kk >> 3) & 1) << 4);
        wsw1[s * W1SZ + ((kt * 8 + (nn >> 4)) * 32 + lan) * 16 + ((kk & 7) + ((kk >> 4) << 3))] =
            (_Float16)w1[nn * 64 + k];
    }
    for (int e = tid; e < 192 * 128; e += nthr) {
        int k = e / 128, nn = e % 128;
        int kt = k >> 5, kk = k & 31;
        int lan = (nn & 15) + (((kk >> 3) & 1) << 4);
        wswp[((kt * 8 + (nn >> 4)) * 32 + lan) * 16 + ((kk & 7) + ((kk >> 4) << 3))] =
            (_Float16)wp[nn * 192 + k];
    }
}

// ---------------------------------------------------------------------------
// Kernel 2: gather + SharedMLP(67->64->128) + BN/ReLU + max-pool via WMMA.
// blockIdx.y = scale. One wave per query; M=16 neighbor tiles.
// Weights async-copied (pre-swizzled f16) into LDS; A operands built
// directly from global feature rows; only layer0->layer1 relayout uses LDS.
// ---------------------------------------------------------------------------
__global__ __launch_bounds__(128) void group_mlp_kernel(
    const float* __restrict__ xyz2, const float* __restrict__ xyz1,
    const _Float16* __restrict__ f1t,
    const int* __restrict__ idx16, const int* __restrict__ idx32,
    const _Float16* __restrict__ wsw0, const _Float16* __restrict__ wsw1,
    const float* __restrict__ g00, const float* __restrict__ b00,
    const float* __restrict__ g01, const float* __restrict__ b01,
    const float* __restrict__ g10, const float* __restrict__ b10,
    const float* __restrict__ g11, const float* __restrict__ b11,
    _Float16* __restrict__ pooled)
{
    __shared__ _Float16 sW0[W0SZ];              // 12 KB
    __shared__ _Float16 sW1[W1SZ];              // 16 KB
    __shared__ _Float16 sA1[4][2 * 32 * 16];    // per-wave act tiles 8 KB

    const int scale = blockIdx.y;
    const float* g0 = scale ? g10 : g00;
    const float* bb0 = scale ? b10 : b00;
    const float* g1 = scale ? g11 : g01;
    const float* bb1 = scale ? b11 : b01;
    const int ns = scale ? 32 : 16;
    const int* idxbuf = scale ? idx32 : idx16;

    // Async-stage pre-swizzled weights into LDS (16B per lane per op).
    {
        const _Float16* src0 = wsw0 + scale * W0SZ;
        const _Float16* src1 = wsw1 + scale * W1SZ;
#pragma unroll
        for (int e = threadIdx.x; e < W0SZ / 8; e += 128)
            async_copy_b128(&sW0[e * 8], src0 + e * 8);
#pragma unroll
        for (int e = threadIdx.x; e < W1SZ / 8; e += 128)
            async_copy_b128(&sW1[e * 8], src1 + e * 8);
        async_wait();
    }
    __syncthreads();

    const int lane = threadIdx.x & 31;
    const int wv = threadIdx.x >> 5;
    const int qi = blockIdx.x * 4 + wv;        // flat b*N2+n2
    const int b = qi >> 11;
    _Float16* A1 = sA1[wv];

    const float qx = xyz2[qi * 3 + 0];
    const float qy = xyz2[qi * 3 + 1];
    const float qz = xyz2[qi * 3 + 2];
    const float* p1 = xyz1 + (size_t)b * N1_ * 3;
    const int n = lane & 15;
    const int g = lane >> 4;
    const float BNK = 0.9999950000374997f;     // 1/sqrt(1+1e-5)

    // Preload BN affine params (channel = nt*16 + n per lane).
    float gs0[4], bt0[4], gs1[8], bt1[8];
#pragma unroll
    for (int nt = 0; nt < 4; ++nt) { gs0[nt] = g0[nt * 16 + n] * BNK; bt0[nt] = bb0[nt * 16 + n]; }
#pragma unroll
    for (int nt = 0; nt < 8; ++nt) { gs1[nt] = g1[nt * 16 + n] * BNK; bt1[nt] = bb1[nt * 16 + n]; }

    float pmax[8];
#pragma unroll
    for (int i = 0; i < 8; ++i) pmax[i] = 0.f;

    const int ntile = ns >> 4;
    for (int t = 0; t < ntile; ++t) {
        const int nb = idxbuf[(size_t)qi * ns + t * 16 + n];

        // ---- build A operands directly from global ----
        const _Float16* frow = f1t + ((size_t)b * N1_ + nb) * 64;
        const uint4* f4 = (const uint4*)frow;        // 8 x 16B chunks
        V16 ua0, ua1;
        ua0.u[0] = f4[0 + g]; ua0.u[1] = f4[2 + g];  // cols [0,32)
        ua1.u[0] = f4[4 + g]; ua1.u[1] = f4[6 + g];  // cols [32,64)
        const float rx = p1[nb * 3 + 0] - qx;
        const float ry = p1[nb * 3 + 1] - qy;
        const float rz = p1[nb * 3 + 2] - qz;
        v16h a2 = {};                                 // cols [64,96): coords+pad
        if (g == 0) {
            a2[0] = (_Float16)rx;                     // c=64
            a2[1] = (_Float16)ry;                     // c=65
            a2[2] = (_Float16)rz;                     // c=66
        }

        // ---- layer 0: (16x96) x (96x64) ----
#pragma unroll
        for (int nt = 0; nt < 4; ++nt) {
            v8f acc = {};
            acc = wmma_f16(ua0.v, ld_op(sW0 + (0 * 4 + nt) * 32 * 16, lane), acc);
            acc = wmma_f16(ua1.v, ld_op(sW0 + (1 * 4 + nt) * 32 * 16, lane), acc);
            acc = wmma_f16(a2,    ld_op(sW0 + (2 * 4 + nt) * 32 * 16, lane), acc);
            // BN+ReLU, scatter into layer-1 A-operand layout
            const int kt1 = nt >> 1;
            const int h = (nt & 1) * 8 + (n & 7);
            _Float16* dst = A1 + ((kt1 * 32 + (((n >> 3) & 1) << 4)) * 16) + h;
#pragma unroll
            for (int r = 0; r < 8; ++r) {
                float v = fmaxf(acc[r] * gs0[nt] + bt0[nt], 0.f);
                dst[(r + 8 * g) * 16] = (_Float16)v;
            }
        }
        lds_fence();

        // ---- layer 1: (16x64) x (64x128), fused max-pool over rows ----
        v16h c0 = ld_op(A1, lane);
        v16h c1 = ld_op(A1 + 32 * 16, lane);
#pragma unroll
        for (int nt = 0; nt < 8; ++nt) {
            v8f acc = {};
            acc = wmma_f16(c0, ld_op(sW1 + (0 * 8 + nt) * 32 * 16, lane), acc);
            acc = wmma_f16(c1, ld_op(sW1 + (1 * 8 + nt) * 32 * 16, lane), acc);
            float mx = pmax[nt];
#pragma unroll
            for (int r = 0; r < 8; ++r)
                mx = fmaxf(mx, fmaxf(acc[r] * gs1[nt] + bt1[nt], 0.f));
            pmax[nt] = mx;
        }
        lds_fence();   // A1 reused next tile
    }

    _Float16* pout = pooled + ((size_t)scale * B_ * N2_ + qi) * 128;
#pragma unroll
    for (int nt = 0; nt < 8; ++nt) {
        float m2 = swap16_max(pmax[nt]);   // combine lane-half row groups
        if (g == 0) pout[nt * 16 + n] = (_Float16)m2;
    }
}

// ---------------------------------------------------------------------------
// Kernel 3: post MLP: concat [pooled(128), f2(64)] -> 128, BN/ReLU via WMMA.
// Fully LDS-free and conversion-free: pooled/f2t are f16 rows; wswp is the
// pre-swizzled f16 B-operand buffer (L2-resident). blockIdx.y = scale.
// ---------------------------------------------------------------------------
__global__ __launch_bounds__(128) void post_mlp_kernel(
    const _Float16* __restrict__ pooled, const _Float16* __restrict__ f2t,
    const _Float16* __restrict__ wswp, const float* __restrict__ gp,
    const float* __restrict__ bp, float* __restrict__ out)
{
    const int scale = blockIdx.y;
    const int lane = threadIdx.x & 31;
    const int wv = threadIdx.x >> 5;
    const int n = lane & 15;
    const int g = lane >> 4;
    const int qbase = (blockIdx.x * 4 + wv) * 16;      // 16 queries per wave
    const int q = qbase + n;                           // this lane's A row

    const _Float16* prow = pooled + ((size_t)scale * B_ * N2_ + q) * 128;
    const _Float16* f2row = f2t + (size_t)q * 64;

    // A operands: kt 0..3 from pooled, kt 4..5 from f2t (all f16 16B chunks).
    v16h a[6];
    {
        const uint4* p4 = (const uint4*)prow;          // 16 x 16B chunks
#pragma unroll
        for (int kt = 0; kt < 4; ++kt) {
            V16 u;
            u.u[0] = p4[kt * 4 + 0 + g];
            u.u[1] = p4[kt * 4 + 2 + g];
            a[kt] = u.v;
        }
        const uint4* f4 = (const uint4*)f2row;         // 8 x 16B chunks
        V16 u4, u5;
        u4.u[0] = f4[0 + g]; u4.u[1] = f4[2 + g];
        u5.u[0] = f4[4 + g]; u5.u[1] = f4[6 + g];
        a[4] = u4.v; a[5] = u5.v;
    }

    const float BNK = 0.9999950000374997f;
#pragma unroll
    for (int nt = 0; nt < 8; ++nt) {
        v8f acc = {};
#pragma unroll
        for (int kt = 0; kt < 6; ++kt)
            acc = wmma_f16(a[kt], ld_op(wswp + ((size_t)(kt * 8 + nt)) * 32 * 16, lane), acc);
        const int nch = nt * 16 + n;
        const float gs = gp[nch] * BNK, bt = bp[nch];
#pragma unroll
        for (int r = 0; r < 8; ++r) {
            const int q2 = qbase + r + 8 * g;
            const int bq = q2 >> 11, n2 = q2 & 2047;
            out[((size_t)bq * 256 + scale * 128 + nch) * N2_ + n2] =
                fmaxf(acc[r] * gs + bt, 0.f);
        }
    }
}

// ---------------------------------------------------------------------------
extern "C" void kernel_launch(void* const* d_in, const int* in_sizes, int n_in,
                              void* d_out, int out_size, void* d_ws, size_t ws_size,
                              hipStream_t stream) {
    const float* xyz2      = (const float*)d_in[0];
    const float* xyz1      = (const float*)d_in[1];
    const float* features2 = (const float*)d_in[2];
    const float* features1 = (const float*)d_in[3];
    const float* w00 = (const float*)d_in[4];
    const float* w01 = (const float*)d_in[5];
    const float* w10 = (const float*)d_in[6];
    const float* w11 = (const float*)d_in[7];
    const float* wp  = (const float*)d_in[8];
    const float* g00 = (const float*)d_in[9];
    const float* b00 = (const float*)d_in[10];
    const float* g01 = (const float*)d_in[11];
    const float* b01 = (const float*)d_in[12];
    const float* g10 = (const float*)d_in[13];
    const float* b10 = (const float*)d_in[14];
    const float* g11 = (const float*)d_in[15];
    const float* b11 = (const float*)d_in[16];
    const float* gp  = (const float*)d_in[17];
    const float* bp  = (const float*)d_in[18];
    float* out = (float*)d_out;

    char* ws = (char*)d_ws;
    int*      idx16  = (int*)(ws);                     //  512 KB
    int*      idx32  = (int*)(ws + 524288);            // 1024 KB
    _Float16* f1t    = (_Float16*)(ws + 1572864);      // 4 MB  (B,N1,C1) f16
    _Float16* f2t    = (_Float16*)(ws + 5767168);      // 1 MB  (B,N2,C2) f16
    _Float16* pooled = (_Float16*)(ws + 6815744);      // 4 MB  2x(B,N2,128) f16
    _Float16* wsw0   = (_Float16*)(ws + 9437184);      // 24 KB swizzled W0 x2
    _Float16* wsw1   = (_Float16*)(ws + 9469952);      // 32 KB swizzled W1 x2
    _Float16* wswp   = (_Float16*)(ws + 9535488);      // 48 KB swizzled Wp

    // 1) ball query (both radii, one ordered wave32 scan)
    bq_kernel<<<dim3(B_ * N2_ / 4), dim3(128), 0, stream>>>(xyz2, xyz1, idx16, idx32);

    // 2) transpose+f16 features; pre-swizzle weights to f16 operand layout
    prep_transpose_kernel<<<dim3(B_ * N1_ * 64 / 256), dim3(256), 0, stream>>>(
        features1, f1t, 64, N1_, B_ * N1_ * 64);
    prep_transpose_kernel<<<dim3(B_ * N2_ * 64 / 256), dim3(256), 0, stream>>>(
        features2, f2t, 64, N2_, B_ * N2_ * 64);
    prep_weights_kernel<<<dim3(64), dim3(256), 0, stream>>>(
        w00, w01, w10, w11, wp, wsw0, wsw1, wswp);

    // 3) grouped MLPs + max-pool (WMMA + async LDS staging), grid.y = scale
    group_mlp_kernel<<<dim3(B_ * N2_ / 4, 2), dim3(128), 0, stream>>>(
        xyz2, xyz1, f1t, idx16, idx32, wsw0, wsw1,
        g00, b00, g01, b01, g10, b10, g11, b11, pooled);

    // 4) post MLP (WMMA, LDS-free, cvt-free), grid.y = scale
    post_mlp_kernel<<<dim3(B_ * N2_ / 64, 2), dim3(128), 0, stream>>>(
        pooled, f2t, wswp, gp, bp, out);
}
